// MultiHeadAttention_37641093382209
// MI455X (gfx1250) — matrix operand order repaired and checked by reading, not compiled, not run on hardware
//
#include <hip/hip_runtime.h>
#include <hip/hip_bf16.h>

#define D_MODEL   1024
#define NUM_HEADS 16
#define HEAD_DIM  64
#define SEQ       2048
#define BATCH     4
#define M_TOTAL   (BATCH * SEQ)   // 8192

typedef __bf16 bf16_t;
typedef __attribute__((ext_vector_type(16))) __bf16 v16bf;
typedef __attribute__((ext_vector_type(8)))  __bf16 v8bf;
typedef __attribute__((ext_vector_type(8)))  float  v8f;

static __device__ __forceinline__ v16bf join8(v8bf lo, v8bf hi) {
  v16bf r;
#pragma unroll
  for (int i = 0; i < 8; ++i) { r[i] = lo[i]; r[i + 8] = hi[i]; }
  return r;
}

static __device__ __forceinline__ v8f wmma_bf16(v16bf a, v16bf b, v8f c) {
  // emits v_wmma_f32_16x16x32_bf16
  return __builtin_amdgcn_wmma_f32_16x16x32_bf16(false, a, false, b, (short)0, c, false, false);
}

// Async copy 16B global -> LDS (per-lane), tracked by ASYNCcnt.
static __device__ __forceinline__ void async_g2l_b128(uint32_t lds_off, uint64_t gaddr) {
  asm volatile("global_load_async_to_lds_b128 %0, %1, off"
               :: "v"(lds_off), "v"(gaddr) : "memory");
}
static __device__ __forceinline__ void wait_asynccnt0() {
  asm volatile("s_wait_asynccnt 0" ::: "memory");
}

// ---------------------------------------------------------------------------
// LayerNorm: one 256-thread block per row of [8192, 1024]; bf16 output.
// ---------------------------------------------------------------------------
__global__ void ln_bf16_kernel(const float* __restrict__ x,
                               const float* __restrict__ gamma,
                               const float* __restrict__ beta,
                               bf16_t* __restrict__ xn) {
  __shared__ float rs[8], rs2[8];
  const int row  = blockIdx.x;
  const int t    = threadIdx.x;
  const int lane = t & 31, w = t >> 5;
  const float* xr = x + (size_t)row * D_MODEL;

  float4 v = *(const float4*)(xr + t * 4);
  float s  = v.x + v.y + v.z + v.w;
  float s2 = v.x * v.x + v.y * v.y + v.z * v.z + v.w * v.w;
#pragma unroll
  for (int off = 16; off; off >>= 1) {
    s  += __shfl_xor(s,  off, 32);
    s2 += __shfl_xor(s2, off, 32);
  }
  if (lane == 0) { rs[w] = s; rs2[w] = s2; }
  __syncthreads();
  if (t == 0) {
    float a = 0.f, b2 = 0.f;
#pragma unroll
    for (int i = 0; i < 8; ++i) { a += rs[i]; b2 += rs2[i]; }
    rs[0] = a; rs2[0] = b2;
  }
  __syncthreads();
  const float mean = rs[0] * (1.0f / D_MODEL);
  const float var  = rs2[0] * (1.0f / D_MODEL) - mean * mean;
  const float rstd = rsqrtf(var + 1e-5f);

  bf16_t* orow = xn + (size_t)row * D_MODEL;
  const int c = t * 4;
  orow[c + 0] = (bf16_t)((v.x - mean) * rstd * gamma[c + 0] + beta[c + 0]);
  orow[c + 1] = (bf16_t)((v.y - mean) * rstd * gamma[c + 1] + beta[c + 1]);
  orow[c + 2] = (bf16_t)((v.z - mean) * rstd * gamma[c + 2] + beta[c + 2]);
  orow[c + 3] = (bf16_t)((v.w - mean) * rstd * gamma[c + 3] + beta[c + 3]);
}

// ---------------------------------------------------------------------------
// fp32 -> bf16 weight downcast.
// ---------------------------------------------------------------------------
__global__ void f2bf_kernel(const float* __restrict__ in, bf16_t* __restrict__ out, int n) {
  int i = blockIdx.x * blockDim.x + threadIdx.x;
  if (i < n) out[i] = (bf16_t)in[i];
}

// ---------------------------------------------------------------------------
// QKV projection GEMM: out[bh,s,hd] = (xn @ W^T + bias) * scale, bf16.
// Block: 128(M) x 64(N); B slab async-staged to LDS per 64-wide K chunk;
// each wave: 16x64 tile, WMMA with A from global, B from LDS.
// ---------------------------------------------------------------------------
__global__ void __launch_bounds__(256)
gemm_qkv_kernel(const bf16_t* __restrict__ A, const bf16_t* __restrict__ W,
                const float* __restrict__ bias, bf16_t* __restrict__ out, float scale) {
  __shared__ __align__(16) bf16_t Bs[64 * 64];   // [n_row][k] 8 KB

  const int lane = threadIdx.x & 31, w = threadIdx.x >> 5;
  const int m0 = blockIdx.y * 128 + w * 16;
  const int n0 = blockIdx.x * 64;
  const int l15    = lane & 15;
  const int c0     = (lane < 16) ? 0 : 8;
  const int kb     = (lane < 16) ? 0 : 16;
  const int rowoff = (lane < 16) ? 0 : 8;

  // async-copy mapping: each thread owns 32B of the 8KB slab
  const int crow = threadIdx.x >> 2;        // 0..63
  const int cseg = (threadIdx.x & 3) * 16;  // element offset 0,16,32,48
  const uint32_t lds0 = (uint32_t)(uintptr_t)(&Bs[crow * 64 + cseg]);
  const uint32_t lds1 = (uint32_t)(uintptr_t)(&Bs[crow * 64 + cseg + 8]);
  const bf16_t* wsrc = W + (size_t)(n0 + crow) * D_MODEL + cseg;

  const bf16_t* arow = A + (size_t)(m0 + l15) * D_MODEL;
  v8f acc[4] = {};

  for (int kk = 0; kk < D_MODEL; kk += 64) {
    __syncthreads();  // protect Bs from previous chunk's readers
    async_g2l_b128(lds0, (uint64_t)(uintptr_t)(wsrc + kk));
    async_g2l_b128(lds1, (uint64_t)(uintptr_t)(wsrc + kk + 8));
    wait_asynccnt0();
    __syncthreads();

#pragma unroll
    for (int kc = 0; kc < 64; kc += 32) {
      __builtin_prefetch(arow + kk + 256, 0, 1);
      v16bf a = join8(*(const v8bf*)(arow + kk + kc + c0),
                      *(const v8bf*)(arow + kk + kc + 16 + c0));
#pragma unroll
      for (int t = 0; t < 4; ++t) {
        const bf16_t* brow = &Bs[(t * 16 + l15) * 64 + kc + c0];
        v16bf b = join8(*(const v8bf*)(brow), *(const v8bf*)(brow + 16));
        acc[t] = wmma_bf16(a, b, acc[t]);
      }
    }
  }

#pragma unroll
  for (int t = 0; t < 4; ++t) {
    const int n = n0 + t * 16 + l15;
    const float bn = bias[n];
    const int h = n >> 6, hd = n & 63;
#pragma unroll
    for (int i = 0; i < 8; ++i) {
      const int m = m0 + rowoff + i;
      const int b = m >> 11, s = m & (SEQ - 1);
      const float vv = (acc[t][i] + bn) * scale;
      out[((size_t)(b * NUM_HEADS + h) * SEQ + s) * HEAD_DIM + hd] = (bf16_t)vv;
    }
  }
}

// ---------------------------------------------------------------------------
// Flash-style causal attention. One wave per 16-query tile of one (b,h).
// KV chunks of 32; scores via 2 WMMAs (chained over Hd=64); online softmax;
// P staged through LDS into A-layout; V staged transposed in LDS; 4 PV WMMAs.
// ---------------------------------------------------------------------------
__global__ void __launch_bounds__(256)
attn_kernel(const bf16_t* __restrict__ Q, const bf16_t* __restrict__ K,
            const bf16_t* __restrict__ V, const float* __restrict__ amask,
            bf16_t* __restrict__ O) {
  __shared__ __align__(16) bf16_t Pbuf[8][16 * 32];
  __shared__ __align__(16) bf16_t Vt[8][64 * 32];

  const int lane = threadIdx.x & 31, w = threadIdx.x >> 5;
  const int q0 = (blockIdx.x * 8 + w) * 16;
  const int bh = blockIdx.y;
  const int b = bh >> 4, h = bh & 15;
  const bf16_t* Qb = Q + (size_t)bh * SEQ * HEAD_DIM;
  const bf16_t* Kb = K + (size_t)bh * SEQ * HEAD_DIM;
  const bf16_t* Vb = V + (size_t)bh * SEQ * HEAD_DIM;

  const int l15    = lane & 15;
  const int c0     = (lane < 16) ? 0 : 8;
  const int kbase  = (lane < 16) ? 0 : 16;
  const int rowoff = (lane < 16) ? 0 : 8;

  // Q tile, A-layout, Hd split into K=0..31 and K=32..63 halves.
  const bf16_t* qrow = Qb + (size_t)(q0 + l15) * HEAD_DIM;
  const v16bf aq0 = join8(*(const v8bf*)(qrow + c0),      *(const v8bf*)(qrow + 16 + c0));
  const v16bf aq1 = join8(*(const v8bf*)(qrow + 32 + c0), *(const v8bf*)(qrow + 48 + c0));

  v8f o[4] = {};
  float mst[8], lst[8];
#pragma unroll
  for (int i = 0; i < 8; ++i) { mst[i] = -1e30f; lst[i] = 0.f; }

  bf16_t* Pw = Pbuf[w];
  bf16_t* Vw = Vt[w];

  for (int k0 = 0; k0 <= q0 + 15; k0 += 32) {
    float p[2][8];
    float cmax[8];
#pragma unroll
    for (int i = 0; i < 8; ++i) cmax[i] = -1e30f;

#pragma unroll
    for (int t = 0; t < 2; ++t) {
      const int key = k0 + t * 16 + l15;
      const bf16_t* krow = Kb + (size_t)key * HEAD_DIM;
      v16bf bk0 = join8(*(const v8bf*)(krow + c0),      *(const v8bf*)(krow + 16 + c0));
      v16bf bk1 = join8(*(const v8bf*)(krow + 32 + c0), *(const v8bf*)(krow + 48 + c0));
      v8f s = {};
      s = wmma_bf16(aq0, bk0, s);
      s = wmma_bf16(aq1, bk1, s);
      const float madd = (1.0f - amask[b * SEQ + key]) * -10000.0f;
#pragma unroll
      for (int i = 0; i < 8; ++i) {
        const int mq = q0 + rowoff + i;
        float sv = s[i] + madd;
        if (key > mq) sv = -1e30f;           // causal mask
        p[t][i] = sv;
        cmax[i] = fmaxf(cmax[i], sv);
      }
    }

    // row max across the 16-lane column group
#pragma unroll
    for (int i = 0; i < 8; ++i) {
      float v = cmax[i];
      v = fmaxf(v, __shfl_xor(v, 1, 32));
      v = fmaxf(v, __shfl_xor(v, 2, 32));
      v = fmaxf(v, __shfl_xor(v, 4, 32));
      v = fmaxf(v, __shfl_xor(v, 8, 32));
      cmax[i] = v;
    }

#pragma unroll
    for (int i = 0; i < 8; ++i) {
      const float mn = fmaxf(mst[i], cmax[i]);
      const float alpha = __expf(mst[i] - mn);
      const float p0 = __expf(p[0][i] - mn);
      const float p1 = __expf(p[1][i] - mn);
      p[0][i] = p0; p[1][i] = p1;
      float rsum = p0 + p1;
      rsum += __shfl_xor(rsum, 1, 32);
      rsum += __shfl_xor(rsum, 2, 32);
      rsum += __shfl_xor(rsum, 4, 32);
      rsum += __shfl_xor(rsum, 8, 32);
      lst[i] = lst[i] * alpha + rsum;
      mst[i] = mn;
#pragma unroll
      for (int t = 0; t < 4; ++t) o[t][i] *= alpha;
    }

    // stage P (C-layout -> LDS row-major [16 x 32])
#pragma unroll
    for (int t = 0; t < 2; ++t)
#pragma unroll
      for (int i = 0; i < 8; ++i)
        Pw[(rowoff + i) * 32 + t * 16 + l15] = (bf16_t)p[t][i];

    // stage V chunk transposed: Vw[hd * 32 + key_local]
    {
      const bf16_t* vrow = Vb + (size_t)(k0 + lane) * HEAD_DIM;
#pragma unroll
      for (int hd = 0; hd < 64; ++hd)
        Vw[hd * 32 + lane] = vrow[hd];
    }
    asm volatile("s_wait_dscnt 0" ::: "memory");

    // P @ V (K-dim = 32 keys)
    const bf16_t* prow = Pw + l15 * 32;
    v16bf ap = join8(*(const v8bf*)(prow + c0), *(const v8bf*)(prow + 16 + c0));
#pragma unroll
    for (int t = 0; t < 4; ++t) {
      const bf16_t* vcol = Vw + (t * 16 + l15) * 32 + c0;
      v16bf bv = join8(*(const v8bf*)(vcol), *(const v8bf*)(vcol + 16));
      o[t] = wmma_bf16(ap, bv, o[t]);
    }
  }

  // normalize + store [B,S,D] bf16
#pragma unroll
  for (int i = 0; i < 8; ++i) {
    const float inv = 1.0f / lst[i];
    const int m = q0 + rowoff + i;
    const size_t base = ((size_t)(b * SEQ + m)) * D_MODEL + h * HEAD_DIM;
#pragma unroll
    for (int t = 0; t < 4; ++t)
      O[base + t * 16 + l15] = (bf16_t)(o[t][i] * inv);
  }
}

// ---------------------------------------------------------------------------
// Output projection + residual: out = AO @ Wo^T + bo + x  (fp32 out)
// Same async-LDS B staging as the QKV GEMM.
// ---------------------------------------------------------------------------
__global__ void __launch_bounds__(256)
gemm_out_kernel(const bf16_t* __restrict__ A, const bf16_t* __restrict__ W,
                const float* __restrict__ bias, const float* __restrict__ resid,
                float* __restrict__ out) {
  __shared__ __align__(16) bf16_t Bs[64 * 64];

  const int lane = threadIdx.x & 31, w = threadIdx.x >> 5;
  const int m0 = blockIdx.y * 128 + w * 16;
  const int n0 = blockIdx.x * 64;
  const int l15    = lane & 15;
  const int c0     = (lane < 16) ? 0 : 8;
  const int kb     = (lane < 16) ? 0 : 16;
  const int rowoff = (lane < 16) ? 0 : 8;

  const int crow = threadIdx.x >> 2;
  const int cseg = (threadIdx.x & 3) * 16;
  const uint32_t lds0 = (uint32_t)(uintptr_t)(&Bs[crow * 64 + cseg]);
  const uint32_t lds1 = (uint32_t)(uintptr_t)(&Bs[crow * 64 + cseg + 8]);
  const bf16_t* wsrc = W + (size_t)(n0 + crow) * D_MODEL + cseg;

  const bf16_t* arow = A + (size_t)(m0 + l15) * D_MODEL;
  v8f acc[4] = {};

  for (int kk = 0; kk < D_MODEL; kk += 64) {
    __syncthreads();
    async_g2l_b128(lds0, (uint64_t)(uintptr_t)(wsrc + kk));
    async_g2l_b128(lds1, (uint64_t)(uintptr_t)(wsrc + kk + 8));
    wait_asynccnt0();
    __syncthreads();

#pragma unroll
    for (int kc = 0; kc < 64; kc += 32) {
      __builtin_prefetch(arow + kk + 256, 0, 1);
      v16bf a = join8(*(const v8bf*)(arow + kk + kc + c0),
                      *(const v8bf*)(arow + kk + kc + 16 + c0));
#pragma unroll
      for (int t = 0; t < 4; ++t) {
        const bf16_t* brow = &Bs[(t * 16 + l15) * 64 + kc + c0];
        v16bf b = join8(*(const v8bf*)(brow), *(const v8bf*)(brow + 16));
        acc[t] = wmma_bf16(a, b, acc[t]);
      }
    }
  }

#pragma unroll
  for (int t = 0; t < 4; ++t) {
    const int n = n0 + t * 16 + l15;
    const float bn = bias[n];
#pragma unroll
    for (int i = 0; i < 8; ++i) {
      const int m = m0 + rowoff + i;
      const size_t idx = (size_t)m * D_MODEL + n;
      out[idx] = acc[t][i] + bn + resid[idx];
    }
  }
}

// ---------------------------------------------------------------------------
extern "C" void kernel_launch(void* const* d_in, const int* in_sizes, int n_in,
                              void* d_out, int out_size, void* d_ws, size_t ws_size,
                              hipStream_t stream) {
  const float* x     = (const float*)d_in[0];
  const float* amask = (const float*)d_in[1];
  const float* Wq    = (const float*)d_in[2];
  const float* bq    = (const float*)d_in[3];
  const float* Wk    = (const float*)d_in[4];
  const float* bk    = (const float*)d_in[5];
  const float* Wv    = (const float*)d_in[6];
  const float* bv    = (const float*)d_in[7];
  const float* Wo    = (const float*)d_in[8];
  const float* bo    = (const float*)d_in[9];
  const float* g     = (const float*)d_in[10];
  const float* be    = (const float*)d_in[11];
  float* out = (float*)d_out;

  char* ws = (char*)d_ws;
  size_t off = 0;
  auto carve = [&](size_t bytes) -> void* {
    void* p = ws + off;
    off += (bytes + 255) & ~(size_t)255;
    return p;
  };
  const size_t actB = (size_t)M_TOTAL * D_MODEL * sizeof(bf16_t); // 16 MB
  const size_t wB   = (size_t)D_MODEL * D_MODEL * sizeof(bf16_t); //  2 MB
  bf16_t* xn  = (bf16_t*)carve(actB);
  bf16_t* Wqb = (bf16_t*)carve(wB);
  bf16_t* Wkb = (bf16_t*)carve(wB);
  bf16_t* Wvb = (bf16_t*)carve(wB);
  bf16_t* Wob = (bf16_t*)carve(wB);
  bf16_t* Qb  = (bf16_t*)carve(actB);
  bf16_t* Kb  = (bf16_t*)carve(actB);
  bf16_t* Vb  = (bf16_t*)carve(actB);
  bf16_t* AO  = (bf16_t*)carve(actB);

  // 1) LayerNorm -> bf16 activations
  ln_bf16_kernel<<<M_TOTAL, 256, 0, stream>>>(x, g, be, xn);

  // 2) weights -> bf16
  const int nW = D_MODEL * D_MODEL;
  f2bf_kernel<<<nW / 256, 256, 0, stream>>>(Wq, Wqb, nW);
  f2bf_kernel<<<nW / 256, 256, 0, stream>>>(Wk, Wkb, nW);
  f2bf_kernel<<<nW / 256, 256, 0, stream>>>(Wv, Wvb, nW);
  f2bf_kernel<<<nW / 256, 256, 0, stream>>>(Wo, Wob, nW);

  // 3) Q/K/V projections (Q pre-scaled by 1/sqrt(Hd))
  dim3 gg(D_MODEL / 64, M_TOTAL / 128);
  gemm_qkv_kernel<<<gg, 256, 0, stream>>>(xn, Wqb, bq, Qb, 0.125f);
  gemm_qkv_kernel<<<gg, 256, 0, stream>>>(xn, Wkb, bk, Kb, 1.0f);
  gemm_qkv_kernel<<<gg, 256, 0, stream>>>(xn, Wvb, bv, Vb, 1.0f);

  // 4) causal flash attention
  attn_kernel<<<dim3(SEQ / 128, BATCH * NUM_HEADS), 256, 0, stream>>>(Qb, Kb, Vb, amask, AO);

  // 5) output projection + bias + residual
  gemm_out_kernel<<<gg, 256, 0, stream>>>(AO, Wob, bo, x, out);

  (void)in_sizes; (void)n_in; (void)out_size; (void)ws_size;
}
